// CCTransformer_7000796692566
// MI455X (gfx1250) — matrix operand order repaired and checked by reading, not compiled, not run on hardware
//
#include <hip/hip_runtime.h>
#include <math.h>

// ---------------------------------------------------------------------------
// Types for CDNA5 WMMA / TDM
// ---------------------------------------------------------------------------
typedef __attribute__((ext_vector_type(16))) __bf16 v16bf;
typedef __attribute__((ext_vector_type(8)))  float  v8f;
typedef __attribute__((ext_vector_type(4)))  unsigned int v4u;
typedef __attribute__((ext_vector_type(8)))  int v8i;
typedef __attribute__((ext_vector_type(4)))  int v4i;

union Frag {
    v16bf v;
    uint4 q[2];
};

__device__ inline unsigned short f2b(float f) {
    unsigned u = __float_as_uint(f);
    unsigned r = u + 0x7FFFu + ((u >> 16) & 1u);   // round-to-nearest-even
    return (unsigned short)(r >> 16);
}

// ---------------------------------------------------------------------------
// Tensor Data Mover: load a 2D bf16 tile (t0 x t1 elements, row-major with
// row stride `stride` elements) from global tile-start `gptr` into LDS at
// byte offset `lds_off`.  d0/d1 are the remaining valid extents from the tile
// start (TDM zero-fills OOB reads).  D# layout per CDNA5 ISA.
// clang-23 builtin arity: (v4u g0, v8i g1, v4i g2, v4i g3, v8i g4, i32 cpol).
// ---------------------------------------------------------------------------
__device__ inline void tdm_load_2d(unsigned lds_off, const unsigned short* gptr,
                                   int d0, int d1, int t0, int t1,
                                   long long stride)
{
    unsigned long long ga = (unsigned long long)(size_t)(const void*)gptr;
    v4u g0;
    g0[0] = 1u;                                       // count=1, user mode
    g0[1] = lds_off;                                  // lds_addr (bytes)
    g0[2] = (unsigned)(ga & 0xFFFFFFFFull);           // global_addr[31:0]
    g0[3] = (unsigned)((ga >> 32) & 0x01FFFFFFull)    // global_addr[56:32]
          | (2u << 30);                               // type=2 ("image")
    unsigned ud0 = (unsigned)d0, ud1 = (unsigned)d1;
    unsigned long long st = (unsigned long long)stride;
    v8i g1;
    g1[0] = (int)(1u << 16);                          // data_size=2 bytes
    g1[1] = (int)((ud0 & 0xFFFFu) << 16);             // tensor_dim0 lo
    g1[2] = (int)(((ud0 >> 16) & 0xFFFFu) | ((ud1 & 0xFFFFu) << 16));
    g1[3] = (int)(((ud1 >> 16) & 0xFFFFu) | (((unsigned)t0 & 0xFFFFu) << 16));
    g1[4] = (int)((unsigned)t1 & 0xFFFFu);            // tile_dim1 (tile_dim2=0)
    g1[5] = (int)(unsigned)(st & 0xFFFFFFFFull);      // dim0 stride lo
    g1[6] = (int)(unsigned)((st >> 32) & 0xFFFFull);  // dim0 stride hi
    g1[7] = 0;
    v4i z4 = { 0, 0, 0, 0 };
    v8i z8 = { 0, 0, 0, 0, 0, 0, 0, 0 };
    __builtin_amdgcn_tensor_load_to_lds(g0, g1, z4, z4, z8, 0);
}

#define TKG 32
#define GTM 64
#define GTN 128

// ---------------------------------------------------------------------------
// Generic GEMM:  C[M,N] = A[M,K] * B[N,K]^T (+ bias[N]) (+ relu)
// A, B bf16 (ushort), C f32.  K % 32 == 0, N % 128 == 0 required.
// Block tile 64(M) x 128(N); 8 waves; wave = 1 A-frag x 4 B-frags = 4 WMMA.
// A/B tiles staged by the Tensor Data Mover, double buffered.
// grid: (N/128, ceil(M/64))
// ---------------------------------------------------------------------------
__global__ void __launch_bounds__(256) cct_gemm(
    const unsigned short* __restrict__ A,
    const unsigned short* __restrict__ B,
    float* __restrict__ C,
    int M, int N, int K,
    const float* __restrict__ bias, int relu)
{
    __shared__ __align__(16) unsigned short As[2][GTM * TKG];
    __shared__ __align__(16) unsigned short Bt[2][GTN * TKG];   // [n][k]

    const int tid  = threadIdx.x;
    const int m0   = blockIdx.y * GTM;
    const int n0   = blockIdx.x * GTN;
    const int wv   = tid >> 5;
    const int lane = tid & 31;
    const int half = lane >> 4;
    const int l16  = lane & 15;
    const int mi   = wv & 3;        // 16-row block
    const int nq   = wv >> 2;       // 64-col group (0/1)

    v8f cacc[4];
    cacc[0] = {}; cacc[1] = {}; cacc[2] = {}; cacc[3] = {};

    unsigned lda[2], ldb[2];
    lda[0] = (unsigned)(size_t)(void*)&As[0][0];
    lda[1] = (unsigned)(size_t)(void*)&As[1][0];
    ldb[0] = (unsigned)(size_t)(void*)&Bt[0][0];
    ldb[1] = (unsigned)(size_t)(void*)&Bt[1][0];

    if (wv == 0) {
        tdm_load_2d(lda[0], A + (size_t)m0 * K, K, M - m0, TKG, GTM, K);
        tdm_load_2d(ldb[0], B + (size_t)n0 * K, K, N - n0, TKG, GTN, K);
    }

    int buf = 0;
    for (int k0 = 0; k0 < K; k0 += TKG) {
        const int nk = k0 + TKG;
        if (wv == 0) {
            if (nk < K) {
                tdm_load_2d(lda[buf ^ 1], A + (size_t)m0 * K + nk, K - nk, M - m0, TKG, GTM, K);
                tdm_load_2d(ldb[buf ^ 1], B + (size_t)n0 * K + nk, K - nk, N - n0, TKG, GTN, K);
                __builtin_amdgcn_s_wait_tensorcnt(2);   // current tile done
            } else {
                __builtin_amdgcn_s_wait_tensorcnt(0);
            }
        }
        __syncthreads();

        const unsigned short* as = &As[buf][0];
        const unsigned short* bt = &Bt[buf][0];
        Frag a;
        a.q[0] = *(const uint4*)(as + (mi * 16 + l16) * TKG + half * 8);
        a.q[1] = *(const uint4*)(as + (mi * 16 + l16) * TKG + 16 + half * 8);
#pragma unroll
        for (int j = 0; j < 4; ++j) {
            int nr = nq * 64 + j * 16 + l16;
            Frag b;
            b.q[0] = *(const uint4*)(bt + nr * TKG + half * 8);
            b.q[1] = *(const uint4*)(bt + nr * TKG + 16 + half * 8);
            cacc[j] = __builtin_amdgcn_wmma_f32_16x16x32_bf16(
                false, a.v, false, b.v, (short)0, cacc[j], false, false);
        }
        __syncthreads();
        buf ^= 1;
    }

#pragma unroll
    for (int j = 0; j < 4; ++j) {
        int col = n0 + nq * 64 + j * 16 + l16;
        if (col < N) {
            float bv = bias ? bias[col] : 0.0f;
#pragma unroll
            for (int i = 0; i < 8; ++i) {
                int row = m0 + mi * 16 + i + half * 8;
                if (row < M) {
                    float v = cacc[j][i] + bv;
                    if (relu) v = fmaxf(v, 0.0f);
                    C[(size_t)row * N + col] = v;
                }
            }
        }
    }
}

// ---------------------------------------------------------------------------
// Implicit-GEMM conv, NCHW, 'SAME' padding, stride 1.
// M = Cout (weights OIHW row-major = (Cout, Cin*KH*KW)),
// N = two image rows (128 pixels, W==64), K = Cin*KH*KW.
// Weight tile staged by TDM; im2col patch gathered manually (padding).
// Channel-concat input: ci < Csplit -> in0, else in1.
// grid: (H/2, ceil(Cout/64), B)
// ---------------------------------------------------------------------------
__global__ void __launch_bounds__(256) cct_conv(
    const unsigned short* __restrict__ in0,
    const unsigned short* __restrict__ in1,
    int Csplit,
    const unsigned short* __restrict__ Wt,
    float* __restrict__ out,
    const float* __restrict__ bias, int relu,
    int Cin, int Cout, int H, int W, int KH, int KW)
{
    __shared__ __align__(16) unsigned short As[GTM * TKG];
    __shared__ __align__(16) unsigned short Bt[GTN * TKG];   // [pixel][k]

    const int tid  = threadIdx.x;
    const int h0   = blockIdx.x * 2;
    const int m0   = blockIdx.y * GTM;
    const int n    = blockIdx.z;
    const int K    = Cin * KH * KW;
    const int wv   = tid >> 5;
    const int lane = tid & 31;
    const int half = lane >> 4;
    const int l16  = lane & 15;
    const int mi   = wv & 3;
    const int nq   = wv >> 2;
    const int PH   = KH >> 1, PW = KW >> 1;
    const size_t plane = (size_t)H * W;

    v8f cacc[4];
    cacc[0] = {}; cacc[1] = {}; cacc[2] = {}; cacc[3] = {};

    const unsigned ldsA = (unsigned)(size_t)(void*)&As[0];

    for (int k0 = 0; k0 < K; k0 += TKG) {
        __syncthreads();
        if (wv == 0)    // weight tile via TDM (overlaps patch gather below)
            tdm_load_2d(ldsA, Wt + (size_t)m0 * K + k0, K - k0, Cout - m0,
                        TKG, GTM, K);
        {   // im2col patch tile: 128 pixels x 32 k
            int p = tid & 127;
            int h = h0 + (p >> 6);
            int w = p & 63;
#pragma unroll
            for (int cchunk = 0; cchunk < 2; ++cchunk) {
                int kb = (tid >> 7) * 16 + cchunk * 8;
                union { unsigned short us[8]; uint4 u4; } tmp;
#pragma unroll
                for (int e = 0; e < 8; ++e) {
                    int kk  = k0 + kb + e;
                    int ci  = kk / (KH * KW);
                    int rem = kk - ci * (KH * KW);
                    int kh  = rem / KW;
                    int kw  = rem - kh * KW;
                    int hh  = h + kh - PH;
                    int ww  = w + kw - PW;
                    unsigned short vs = 0;
                    if (hh >= 0 && hh < H && ww >= 0 && ww < W) {
                        const unsigned short* sp = (ci < Csplit)
                            ? in0 + ((size_t)n * Csplit + ci) * plane
                            : in1 + ((size_t)n * (Cin - Csplit) + (ci - Csplit)) * plane;
                        vs = sp[(size_t)hh * W + ww];
                    }
                    tmp.us[e] = vs;
                }
                *(uint4*)(Bt + p * TKG + kb) = tmp.u4;
            }
        }
        if (wv == 0) __builtin_amdgcn_s_wait_tensorcnt(0);
        __syncthreads();

        Frag a;
        a.q[0] = *(const uint4*)(As + (mi * 16 + l16) * TKG + half * 8);
        a.q[1] = *(const uint4*)(As + (mi * 16 + l16) * TKG + 16 + half * 8);
#pragma unroll
        for (int j = 0; j < 4; ++j) {
            int nr = nq * 64 + j * 16 + l16;
            Frag b;
            b.q[0] = *(const uint4*)(Bt + nr * TKG + half * 8);
            b.q[1] = *(const uint4*)(Bt + nr * TKG + 16 + half * 8);
            cacc[j] = __builtin_amdgcn_wmma_f32_16x16x32_bf16(
                false, a.v, false, b.v, (short)0, cacc[j], false, false);
        }
    }

#pragma unroll
    for (int j = 0; j < 4; ++j) {
        int p = nq * 64 + j * 16 + l16;      // pixel within the 2-row tile
        int h = h0 + (p >> 6);
        int w = p & 63;
#pragma unroll
        for (int i = 0; i < 8; ++i) {
            int co = m0 + mi * 16 + i + half * 8;
            if (co < Cout) {
                float v = cacc[j][i] + (bias ? bias[co] : 0.0f);
                if (relu) v = fmaxf(v, 0.0f);
                out[((size_t)n * Cout + co) * plane + (size_t)h * W + w] = v;
            }
        }
    }
}

// ---------------------------------------------------------------------------
// Criss-cross attention core (per-pixel): scores over H column + W row,
// softmax over 128, aggregate V, out = gamma*agg + x.
// grid: (W, H, B), block 128.
// ---------------------------------------------------------------------------
__global__ void __launch_bounds__(128) cct_cc(
    const float* __restrict__ q, const float* __restrict__ k,
    const float* __restrict__ v, const float* __restrict__ x,
    const float* __restrict__ gamma_p, int gidx,
    float* __restrict__ out, int B, int C, int Cq, int H, int W)
{
    const int w = blockIdx.x, h = blockIdx.y, b = blockIdx.z;
    const int tid = threadIdx.x;
    __shared__ float qs[8];
    __shared__ float sc[128];
    __shared__ float red[128];
    const size_t plane = (size_t)H * W;

    if (tid < Cq)
        qs[tid] = q[((size_t)b * Cq + tid) * plane + (size_t)h * W + w];
    __syncthreads();

    float s = 0.0f;
    if (tid < H) {
        int j = tid;
        for (int c = 0; c < Cq; ++c)
            s += qs[c] * k[((size_t)b * Cq + c) * plane + (size_t)j * W + w];
        if (j == h) s += -1.0e9f;
    } else {
        int j = tid - H;
        for (int c = 0; c < Cq; ++c)
            s += qs[c] * k[((size_t)b * Cq + c) * plane + (size_t)h * W + j];
    }
    sc[tid] = s;
    red[tid] = s;
    __syncthreads();
    for (int off = 64; off > 0; off >>= 1) {
        if (tid < off) red[tid] = fmaxf(red[tid], red[tid + off]);
        __syncthreads();
    }
    float mx = red[0];
    __syncthreads();
    float e = __expf(s - mx);
    sc[tid] = e;
    red[tid] = e;
    __syncthreads();
    for (int off = 64; off > 0; off >>= 1) {
        if (tid < off) red[tid] += red[tid + off];
        __syncthreads();
    }
    float inv = 1.0f / red[0];

    if (tid < C) {
        int c = tid;
        const float* vb = v + ((size_t)b * C + c) * plane;
        float acc = 0.0f;
        for (int j = 0; j < H; ++j) acc += vb[(size_t)j * W + w] * sc[j];
        for (int j = 0; j < W; ++j) acc += vb[(size_t)h * W + j] * sc[H + j];
        size_t idx = ((size_t)b * C + c) * plane + (size_t)h * W + w;
        out[idx] = gamma_p[gidx] * (acc * inv) + x[idx];
    }
}

// ---------------------------------------------------------------------------
// MHA attention core: one block per (query l, head hd, batch b).
// K/V row for key j is (j*B + b). dh <= 64. dyn-shared = (S+128) floats.
// ---------------------------------------------------------------------------
__global__ void __launch_bounds__(128) cct_attn(
    const float* __restrict__ Xq, int ldq, int qoff,
    const float* __restrict__ Xk, int ldk, int koff,
    const float* __restrict__ Xv, int ldv, int voff,
    float* __restrict__ Out, int ldo,
    int B, int S, int dh, float scale,
    const unsigned char* __restrict__ mask)
{
    const int l = blockIdx.x, hd = blockIdx.y, b = blockIdx.z;
    const int tid = threadIdx.x;
    __shared__ float qs[64];
    extern __shared__ float sc[];
    float* red = sc + S;
    const int qrow = l * B + b;

    if (tid < dh)
        qs[tid] = Xq[(size_t)qrow * ldq + qoff + hd * dh + tid];
    __syncthreads();

    float lmax = -3.0e38f;
    for (int j = tid; j < S; j += 128) {
        const float* kp = Xk + (size_t)(j * B + b) * ldk + koff + hd * dh;
        float acc = 0.0f;
        for (int d = 0; d < dh; ++d) acc += qs[d] * kp[d];
        acc *= scale;
        if (mask && mask[(size_t)b * S + j]) acc = -1.0e9f;
        sc[j] = acc;
        lmax = fmaxf(lmax, acc);
    }
    red[tid] = lmax;
    __syncthreads();
    for (int off = 64; off > 0; off >>= 1) {
        if (tid < off) red[tid] = fmaxf(red[tid], red[tid + off]);
        __syncthreads();
    }
    float mx = red[0];
    __syncthreads();
    float lsum = 0.0f;
    for (int j = tid; j < S; j += 128) {
        float e = __expf(sc[j] - mx);
        sc[j] = e;
        lsum += e;
    }
    red[tid] = lsum;
    __syncthreads();
    for (int off = 64; off > 0; off >>= 1) {
        if (tid < off) red[tid] += red[tid + off];
        __syncthreads();
    }
    float inv = 1.0f / red[0];

    if (tid < dh) {
        int d = tid;
        float acc = 0.0f;
        for (int j = 0; j < S; ++j)
            acc += sc[j] * Xv[(size_t)(j * B + b) * ldv + voff + hd * dh + d];
        Out[(size_t)qrow * ldo + hd * dh + d] = acc * inv;
    }
}

// ---------------------------------------------------------------------------
// Fused residual-add + LayerNorm over last dim C (==blockDim.x==256).
// out = LN(a + bsrc) * g + beta ; bsrc may be null. out may alias a.
// ---------------------------------------------------------------------------
__global__ void __launch_bounds__(256) cct_ln(
    const float* __restrict__ a, const float* __restrict__ bsrc,
    const float* __restrict__ g, const float* __restrict__ beta,
    float* __restrict__ out, int C)
{
    const int row = blockIdx.x;
    const int tid = threadIdx.x;
    __shared__ float red[256];
    float x = a[(size_t)row * C + tid] + (bsrc ? bsrc[(size_t)row * C + tid] : 0.0f);
    red[tid] = x;
    __syncthreads();
    for (int off = 128; off > 0; off >>= 1) {
        if (tid < off) red[tid] += red[tid + off];
        __syncthreads();
    }
    float mu = red[0] / (float)C;
    __syncthreads();
    float d = x - mu;
    red[tid] = d * d;
    __syncthreads();
    for (int off = 128; off > 0; off >>= 1) {
        if (tid < off) red[tid] += red[tid + off];
        __syncthreads();
    }
    float var = red[0] / (float)C;
    out[(size_t)row * C + tid] = d * rsqrtf(var + 1e-5f) * g[tid] + beta[tid];
}

// ---------------------------------------------------------------------------
// Elementwise helpers
// ---------------------------------------------------------------------------
__global__ void cct_add(float* __restrict__ o, const float* __restrict__ a,
                        const float* __restrict__ b, size_t n)
{
    size_t i = blockIdx.x * (size_t)blockDim.x + threadIdx.x;
    if (i < n) o[i] = a[i] + b[i];
}

__global__ void cct_f2b(unsigned short* __restrict__ o,
                        const float* __restrict__ a, size_t n)
{
    size_t i = blockIdx.x * (size_t)blockDim.x + threadIdx.x;
    if (i < n) o[i] = f2b(a[i]);
}

__global__ void cct_fill(float* __restrict__ o, float v, size_t n)
{
    size_t i = blockIdx.x * (size_t)blockDim.x + threadIdx.x;
    if (i < n) o[i] = v;
}

// tok[(l*B+b)*C + c] = src[(l*B+b)*C + c] + qe[l*C + c]
__global__ void cct_addqe(float* __restrict__ dst, const float* __restrict__ src,
                          const float* __restrict__ qe, int NQ, int B, int C)
{
    int i = blockIdx.x * blockDim.x + threadIdx.x;
    int total = NQ * B * C;
    if (i < total) {
        int c = i % C;
        int lb = i / C;
        int l = lb / B;
        dst[i] = src[i] + qe[l * C + c];
    }
}

// memtok_bf[(p*B+b)*C + c] = bf16(mem[b][c][p] + pos[b][c][p])
__global__ void cct_memtok(unsigned short* __restrict__ dst,
                           const float* __restrict__ mem,
                           const float* __restrict__ pos,
                           int B, int C, int P)
{
    size_t i = blockIdx.x * (size_t)blockDim.x + threadIdx.x;
    size_t total = (size_t)P * B * C;
    if (i < total) {
        int c = (int)(i % C);
        size_t t = i / C;
        int b = (int)(t % B);
        int p = (int)(t / B);
        size_t src = ((size_t)b * C + c) * P + p;
        dst[i] = f2b(mem[src] + pos[src]);
    }
}

// hs[0][b][l][c] = fin[(l*B+b)*C + c]
__global__ void cct_write_hs(float* __restrict__ o, const float* __restrict__ fin,
                             int NQ, int B, int C)
{
    int i = blockIdx.x * blockDim.x + threadIdx.x;
    int total = NQ * B * C;
    if (i < total) {
        int c = i % C;
        int lb = i / C;
        int b = lb % B;
        int l = lb / B;
        o[((size_t)b * NQ + l) * C + c] = fin[i];
    }
}

// ---------------------------------------------------------------------------
// Host orchestration
// ---------------------------------------------------------------------------
extern "C" void kernel_launch(void* const* d_in, const int* in_sizes, int n_in,
                              void* d_out, int out_size, void* d_ws, size_t ws_size,
                              hipStream_t stream)
{
    (void)in_sizes; (void)n_in; (void)out_size; (void)ws_size;

    const int B = 4, C = 256, H = 64, W = 64, HW = 64 * 64;
    const int L = 6, INTER = 64, C8 = 8, DFF = 2048, NQ = 100, DH = 32, NH = 8;
    const int T = NQ * B;                 // 400 decoder token rows
    const float scale = 1.0f / sqrtf((float)DH);

    // ---- inputs ----
    const float* src     = (const float*)d_in[0];
    const unsigned char* maskp = (const unsigned char*)d_in[1];
    const float* qe      = (const float*)d_in[2];
    const float* pos     = (const float*)d_in[3];
    const float* conva_w = (const float*)d_in[4];
    const float* qw      = (const float*)d_in[5];
    const float* qb      = (const float*)d_in[6];
    const float* kw      = (const float*)d_in[7];
    const float* kb      = (const float*)d_in[8];
    const float* vw      = (const float*)d_in[9];
    const float* vb      = (const float*)d_in[10];
    const float* gammap  = (const float*)d_in[11];
    const float* convb_w = (const float*)d_in[12];
    const float* bneck_w = (const float*)d_in[13];
    const float* saw     = (const float*)d_in[14];
    const float* sab     = (const float*)d_in[15];
    const float* saow    = (const float*)d_in[16];
    const float* saob    = (const float*)d_in[17];
    const float* caw     = (const float*)d_in[18];
    const float* cab     = (const float*)d_in[19];
    const float* caow    = (const float*)d_in[20];
    const float* caob    = (const float*)d_in[21];
    const float* l1w     = (const float*)d_in[22];
    const float* l1b     = (const float*)d_in[23];
    const float* l2w     = (const float*)d_in[24];
    const float* l2b     = (const float*)d_in[25];
    const float* n1g     = (const float*)d_in[26];
    const float* n1b     = (const float*)d_in[27];
    const float* n2g     = (const float*)d_in[28];
    const float* n2b     = (const float*)d_in[29];
    const float* n3g     = (const float*)d_in[30];
    const float* n3b     = (const float*)d_in[31];
    const float* ng      = (const float*)d_in[32];
    const float* nb      = (const float*)d_in[33];

    float* out_f = (float*)d_out;

    // ---- workspace layout ----
    const size_t S_MEM = (size_t)B * C * HW;      // 4194304
    const size_t S_Y   = (size_t)B * INTER * HW;  // 1048576
    const size_t S_QK  = (size_t)B * C8 * HW;     // 131072
    const size_t S_WB  = (size_t)C * 2 * C * 9;   // 1179648 (max weight)
    const size_t S_MT  = (size_t)HW * B * C;      // 4194304
    const size_t S_XKV = (size_t)HW * B * 2 * C;  // 8388608
    const size_t S_T   = (size_t)T * C;           // 102400
    const size_t S_QKX = (size_t)T * 2 * C;       // 204800
    const size_t S_FFN = (size_t)T * DFF;         // 819200

    char* p = (char*)d_ws;
    auto alloc_f = [&](size_t n) { float* r = (float*)p; p += ((n * 4 + 255) / 256) * 256; return r; };
    auto alloc_h = [&](size_t n) { unsigned short* r = (unsigned short*)p; p += ((n * 2 + 255) / 256) * 256; return r; };

    float*          memb = alloc_f(S_MEM);
    float*          xi   = alloc_f(S_MEM);
    unsigned short* xib  = alloc_h(S_MEM);
    float*          z    = alloc_f(S_MEM);
    unsigned short* zb   = alloc_h(S_MEM);
    float*          y    = alloc_f(S_Y);
    float*          y2   = alloc_f(S_Y);
    unsigned short* yb   = alloc_h(S_Y);
    float*          qbuf = alloc_f(S_QK);
    float*          kbuf = alloc_f(S_QK);
    float*          vbuf = alloc_f(S_Y);
    unsigned short* wbuf = alloc_h(S_WB);
    unsigned short* mtb  = alloc_h(S_MT);
    float*          xkv  = alloc_f(S_XKV);
    float*          outd = alloc_f(S_T);
    float*          toka = alloc_f(S_T);
    unsigned short* tokab= alloc_h(S_T);
    unsigned short* outb = alloc_h(S_T);
    float*          xqk  = alloc_f(S_QKX);
    float*          xv   = alloc_f(S_T);
    float*          att  = alloc_f(S_T);
    float*          t2   = alloc_f(S_T);
    float*          xq   = alloc_f(S_T);
    float*          ffn  = alloc_f(S_FFN);
    unsigned short* ffnb = alloc_h(S_FFN);

    auto g1 = [](size_t n) { return dim3((unsigned)((n + 255) / 256)); };

    // mem = src
    (void)hipMemcpyAsync(memb, src, S_MEM * sizeof(float), hipMemcpyDeviceToDevice, stream);

    // =========================== Encoder ===========================
    for (int i = 0; i < L; ++i) {
        // xi = mem + pos
        cct_add<<<g1(S_MEM), 256, 0, stream>>>(xi, memb, pos, S_MEM);
        cct_f2b<<<g1(S_MEM), 256, 0, stream>>>(xib, xi, S_MEM);

        // y = relu(conv3x3(xi, conva_w[i]))   256 -> 64
        cct_f2b<<<g1((size_t)INTER * C * 9), 256, 0, stream>>>(
            wbuf, conva_w + (size_t)i * INTER * C * 9, (size_t)INTER * C * 9);
        cct_conv<<<dim3(H / 2, 1, B), 256, 0, stream>>>(
            xib, (const unsigned short*)nullptr, C, wbuf, y,
            (const float*)nullptr, 1, C, INTER, H, W, 3, 3);

        // criss-cross attention, applied twice
        float* cin = y;
        float* cout2 = y2;
        for (int rep = 0; rep < 2; ++rep) {
            cct_f2b<<<g1(S_Y), 256, 0, stream>>>(yb, cin, S_Y);
            // q = conv1x1 (64->8) + bias
            cct_f2b<<<g1((size_t)C8 * INTER), 256, 0, stream>>>(
                wbuf, qw + (size_t)i * C8 * INTER, (size_t)C8 * INTER);
            cct_conv<<<dim3(H / 2, 1, B), 256, 0, stream>>>(
                yb, (const unsigned short*)nullptr, INTER, wbuf, qbuf,
                qb + (size_t)i * C8, 0, INTER, C8, H, W, 1, 1);
            // k = conv1x1 (64->8) + bias
            cct_f2b<<<g1((size_t)C8 * INTER), 256, 0, stream>>>(
                wbuf, kw + (size_t)i * C8 * INTER, (size_t)C8 * INTER);
            cct_conv<<<dim3(H / 2, 1, B), 256, 0, stream>>>(
                yb, (const unsigned short*)nullptr, INTER, wbuf, kbuf,
                kb + (size_t)i * C8, 0, INTER, C8, H, W, 1, 1);
            // v = conv1x1 (64->64) + bias
            cct_f2b<<<g1((size_t)INTER * INTER), 256, 0, stream>>>(
                wbuf, vw + (size_t)i * INTER * INTER, (size_t)INTER * INTER);
            cct_conv<<<dim3(H / 2, 1, B), 256, 0, stream>>>(
                yb, (const unsigned short*)nullptr, INTER, wbuf, vbuf,
                vb + (size_t)i * INTER, 0, INTER, INTER, H, W, 1, 1);
            // attention core
            cct_cc<<<dim3(W, H, B), 128, 0, stream>>>(
                qbuf, kbuf, vbuf, cin, gammap, i, cout2, B, INTER, C8, H, W);
            float* tmp = cin; cin = cout2; cout2 = tmp;
        }
        // result is in cin (== y after two swaps)

        // z = relu(conv3x3(cin, convb_w[i]))   64 -> 256
        cct_f2b<<<g1(S_Y), 256, 0, stream>>>(yb, cin, S_Y);
        cct_f2b<<<g1((size_t)C * INTER * 9), 256, 0, stream>>>(
            wbuf, convb_w + (size_t)i * C * INTER * 9, (size_t)C * INTER * 9);
        cct_conv<<<dim3(H / 2, C / 64, B), 256, 0, stream>>>(
            yb, (const unsigned short*)nullptr, INTER, wbuf, z,
            (const float*)nullptr, 1, INTER, C, H, W, 3, 3);
        cct_f2b<<<g1(S_MEM), 256, 0, stream>>>(zb, z, S_MEM);

        // mem = conv3x3(concat(xi, z), bneck_w[i])   512 -> 256
        cct_f2b<<<g1((size_t)C * 2 * C * 9), 256, 0, stream>>>(
            wbuf, bneck_w + (size_t)i * C * 2 * C * 9, (size_t)C * 2 * C * 9);
        cct_conv<<<dim3(H / 2, C / 64, B), 256, 0, stream>>>(
            xib, zb, C, wbuf, memb, (const float*)nullptr, 0, 2 * C, C, H, W, 3, 3);
    }

    // =========================== Decoder ===========================
    // memory tokens (memory + pos) in bf16, token row = p*B + b
    cct_memtok<<<g1(S_MT), 256, 0, stream>>>(mtb, memb, pos, B, C, HW);
    cct_fill<<<g1(S_T), 256, 0, stream>>>(outd, 0.0f, S_T);

    const int gT = (T + 63) / 64;   // M tiles for token GEMMs

    for (int i = 0; i < L; ++i) {
        // ---- self attention ----
        cct_addqe<<<g1(S_T), 256, 0, stream>>>(toka, outd, qe, NQ, B, C);
        cct_f2b<<<g1(S_T), 256, 0, stream>>>(tokab, toka, S_T);
        cct_f2b<<<g1(S_T), 256, 0, stream>>>(outb, outd, S_T);
        // Xqk = (out+qe) @ [wq;wk]^T + b[0:512]
        cct_f2b<<<g1((size_t)512 * C), 256, 0, stream>>>(
            wbuf, saw + (size_t)i * 3 * C * C, (size_t)512 * C);
        cct_gemm<<<dim3(512 / 128, gT), 256, 0, stream>>>(
            tokab, wbuf, xqk, T, 512, C, sab + (size_t)i * 3 * C, 0);
        // Xv = out @ wv^T + b[512:768]
        cct_f2b<<<g1((size_t)C * C), 256, 0, stream>>>(
            wbuf, saw + (size_t)i * 3 * C * C + (size_t)512 * C, (size_t)C * C);
        cct_gemm<<<dim3(C / 128, gT), 256, 0, stream>>>(
            outb, wbuf, xv, T, C, C, sab + (size_t)i * 3 * C + 512, 0);
        cct_attn<<<dim3(NQ, NH, B), 128, (NQ + 128) * sizeof(float), stream>>>(
            xqk, 512, 0, xqk, 512, 256, xv, 256, 0, att, 256,
            B, NQ, DH, scale, (const unsigned char*)nullptr);
        // out-proj
        cct_f2b<<<g1(S_T), 256, 0, stream>>>(tokab, att, S_T);
        cct_f2b<<<g1((size_t)C * C), 256, 0, stream>>>(
            wbuf, saow + (size_t)i * C * C, (size_t)C * C);
        cct_gemm<<<dim3(C / 128, gT), 256, 0, stream>>>(
            tokab, wbuf, t2, T, C, C, saob + (size_t)i * C, 0);
        cct_ln<<<T, 256, 0, stream>>>(outd, t2, n1g + (size_t)i * C, n1b + (size_t)i * C, outd, C);

        // ---- cross attention ----
        cct_addqe<<<g1(S_T), 256, 0, stream>>>(toka, outd, qe, NQ, B, C);
        cct_f2b<<<g1(S_T), 256, 0, stream>>>(tokab, toka, S_T);
        // Xq = (out+qe) @ wq^T + b[0:256]
        cct_f2b<<<g1((size_t)C * C), 256, 0, stream>>>(
            wbuf, caw + (size_t)i * 3 * C * C, (size_t)C * C);
        cct_gemm<<<dim3(C / 128, gT), 256, 0, stream>>>(
            tokab, wbuf, xq, T, C, C, cab + (size_t)i * 3 * C, 0);
        // Xkv = memtok @ [wk;wv]^T + b[256:768]
        cct_f2b<<<g1((size_t)512 * C), 256, 0, stream>>>(
            wbuf, caw + (size_t)i * 3 * C * C + (size_t)C * C, (size_t)512 * C);
        cct_gemm<<<dim3(512 / 128, (HW * B) / 64), 256, 0, stream>>>(
            mtb, wbuf, xkv, HW * B, 512, C, cab + (size_t)i * 3 * C + C, 0);
        cct_attn<<<dim3(NQ, NH, B), 128, (HW + 128) * sizeof(float), stream>>>(
            xq, 256, 0, xkv, 512, 0, xkv, 512, 256, att, 256,
            B, HW, DH, scale, maskp);
        // out-proj
        cct_f2b<<<g1(S_T), 256, 0, stream>>>(tokab, att, S_T);
        cct_f2b<<<g1((size_t)C * C), 256, 0, stream>>>(
            wbuf, caow + (size_t)i * C * C, (size_t)C * C);
        cct_gemm<<<dim3(C / 128, gT), 256, 0, stream>>>(
            tokab, wbuf, t2, T, C, C, caob + (size_t)i * C, 0);
        cct_ln<<<T, 256, 0, stream>>>(outd, t2, n2g + (size_t)i * C, n2b + (size_t)i * C, outd, C);

        // ---- FFN ----
        cct_f2b<<<g1(S_T), 256, 0, stream>>>(outb, outd, S_T);
        cct_f2b<<<g1((size_t)DFF * C), 256, 0, stream>>>(
            wbuf, l1w + (size_t)i * DFF * C, (size_t)DFF * C);
        cct_gemm<<<dim3(DFF / 128, gT), 256, 0, stream>>>(
            outb, wbuf, ffn, T, DFF, C, l1b + (size_t)i * DFF, 1);
        cct_f2b<<<g1(S_FFN), 256, 0, stream>>>(ffnb, ffn, S_FFN);
        cct_f2b<<<g1((size_t)C * DFF), 256, 0, stream>>>(
            wbuf, l2w + (size_t)i * C * DFF, (size_t)C * DFF);
        cct_gemm<<<dim3(C / 128, gT), 256, 0, stream>>>(
            ffnb, wbuf, t2, T, C, DFF, l2b + (size_t)i * C, 0);
        cct_ln<<<T, 256, 0, stream>>>(outd, t2, n3g + (size_t)i * C, n3b + (size_t)i * C, outd, C);
    }

    // final layer norm -> toka
    cct_ln<<<T, 256, 0, stream>>>(outd, (const float*)nullptr, ng, nb, toka, C);

    // outputs: hs (1,B,NQ,C) then memory_2d (B,C,H,W)
    cct_write_hs<<<g1(S_T), 256, 0, stream>>>(out_f, toka, NQ, B, C);
    (void)hipMemcpyAsync(out_f + (size_t)NQ * B * C, memb, S_MEM * sizeof(float),
                         hipMemcpyDeviceToDevice, stream);
}
